// RandmanSNN_74878459839021
// MI455X (gfx1250) — matrix operand order repaired and checked by reading, not compiled, not run on hardware
//
#include <hip/hip_runtime.h>

// ---------------------------------------------------------------------------
// SNN forward for MI455X (gfx1250, wave32, WMMA + async-LDS + TDM).
//   k0: W2 fp32 -> f16 into d_ws (async-path only)
//   k1: fused GEMM1 (x@W1^T, f16 WMMA, f32 acc) + LIF1 -> spk1_rec (f32)
//       and optionally f16 spike copy into d_ws
//   k2: GEMM2 (spk1@W2^T) -> cur2 into mem2_rec region
//       async variant: W2 tile via TENSOR_LOAD_TO_LDS (TDM, TENSORcnt),
//       spike rows via global_load_async_to_lds_b128 (ASYNCcnt)
//   k3: in-place beta2 prefix scan over mem2_rec
// ---------------------------------------------------------------------------

typedef _Float16 v16h __attribute__((ext_vector_type(16)));
typedef _Float16 v8h  __attribute__((ext_vector_type(8)));
typedef _Float16 v4h  __attribute__((ext_vector_type(4)));
typedef float    v8f  __attribute__((ext_vector_type(8)));
typedef unsigned int v4u __attribute__((ext_vector_type(4)));
typedef int          v8i __attribute__((ext_vector_type(8)));
typedef int          v4i __attribute__((ext_vector_type(4)));

#define SNN_B  256
#define SNN_T  128
#define SNN_NI 512
#define SNN_NH 1024
#define SNN_NO 64

// +8 halves (16B) row padding: row stride in dwords == 4 mod 64 banks
#define XS_STRIDE  (SNN_NI + 8)   // 520
#define W1S_STRIDE (SNN_NI + 8)   // 520
#define NHS_STRIDE (SNN_NH + 8)   // 1032
// TDM-compatible W2 layout: pad 4 dwords after every 256 dwords
// -> row stride 1040 halves, +8 half bump at half-column 512.
#define W2T_STRIDE 1040

static __device__ __forceinline__ v16h cat8(v8h lo, v8h hi) {
  return __builtin_shufflevector(lo, hi, 0, 1, 2, 3, 4, 5, 6, 7,
                                         8, 9, 10, 11, 12, 13, 14, 15);
}

// Compile-time-unrolled async copy: each lane moves END contiguous bytes
// from (base + goff) to LDS byte offset loff, 16B per instruction.
// The offset: immediate applies to BOTH the global and LDS addresses.
template <int J, int END>
struct AsyncCp {
  static __device__ __forceinline__ void go(unsigned loff, unsigned goff,
                                            const void* base) {
    asm volatile("global_load_async_to_lds_b128 %0, %1, %2 offset:%3"
                 :: "v"(loff), "v"(goff), "s"(base), "n"(J) : "memory");
    AsyncCp<J + 16, END>::go(loff, goff, base);
  }
};
template <int END>
struct AsyncCp<END, END> {
  static __device__ __forceinline__ void go(unsigned, unsigned, const void*) {}
};

static __device__ __forceinline__ void wait_asynccnt0() {
  asm volatile("s_wait_asynccnt 0x0" ::: "memory");
}

// -------------------------------- kernel 0 ---------------------------------
__global__ void __launch_bounds__(256)
snn_w2_to_f16(const float* __restrict__ W2, _Float16* __restrict__ w2h) {
  const int i4 = (blockIdx.x * 256 + threadIdx.x) * 4;  // grid covers NO*NH
  float4 v = *(const float4*)(W2 + i4);
  v4h h = {(_Float16)v.x, (_Float16)v.y, (_Float16)v.z, (_Float16)v.w};
  *(v4h*)(w2h + i4) = h;
}

// -------------------------------- kernel 1 ---------------------------------
// Block: 256 threads = 8 waves. Tile: 16 batch rows x 128 hidden cols.
// Grid: (B/16, NH/128). mem1 lives in the 8 accumulator VGPRs.
__global__ void __launch_bounds__(256)
snn_fc1_lif1(const float* __restrict__ x, const float* __restrict__ W1,
             const float* __restrict__ beta1p, float* __restrict__ spk_out,
             _Float16* __restrict__ spkh_out, int emit_f16) {
  extern __shared__ _Float16 smem1[];
  _Float16* w1s = smem1;                      // [128][W1S_STRIDE] f16
  _Float16* xs  = smem1 + 128 * W1S_STRIDE;   // [16][XS_STRIDE]   f16

  const int tid = threadIdx.x;
  const int b0  = blockIdx.x * 16;
  const int h0  = blockIdx.y * 128;
  const float beta1 = *beta1p;

  {
    const int row  = tid >> 1;            // 0..127
    const int col0 = (tid & 1) * 256;
    const float* src = W1 + (size_t)(h0 + row) * SNN_NI + col0;
    _Float16* dst = w1s + row * W1S_STRIDE + col0;
#pragma unroll 8
    for (int c = 0; c < 256; c += 4) {
      float4 v = *(const float4*)(src + c);
      v4h h = {(_Float16)v.x, (_Float16)v.y, (_Float16)v.z, (_Float16)v.w};
      *(v4h*)(dst + c) = h;
    }
  }

  const int wave = tid >> 5;
  const int lane = tid & 31;
  const int hi16 = lane >> 4;
  const int l16  = lane & 15;

  v8f mem = {};  // mem1 state for this wave's 16x16 tile (8 elems/lane)

  const _Float16* aBase = xs  + l16 * XS_STRIDE + hi16 * 16;
  const _Float16* bBase = w1s + (wave * 16 + l16) * W1S_STRIDE + hi16 * 16;

  for (int t = 0; t < SNN_T; ++t) {
    __syncthreads();  // W1 ready (t==0) / previous x tile fully consumed
    {
      const int row  = tid >> 4;           // 0..15
      const int col0 = (tid & 15) * 32;
      const float* src = x + ((size_t)(b0 + row) * SNN_T + t) * SNN_NI + col0;
      _Float16* dst = xs + row * XS_STRIDE + col0;
#pragma unroll
      for (int c = 0; c < 32; c += 4) {
        float4 v = *(const float4*)(src + c);
        v4h h = {(_Float16)v.x, (_Float16)v.y, (_Float16)v.z, (_Float16)v.w};
        *(v4h*)(dst + c) = h;
      }
      if (t + 1 < SNN_T) __builtin_prefetch(src + SNN_NI, 0, 0);  // next t
    }
    __syncthreads();

    v8f acc = {};
#pragma unroll
    for (int k0 = 0; k0 < SNN_NI; k0 += 32) {
      v8h alo = *(const v8h*)(aBase + k0);
      v8h ahi = *(const v8h*)(aBase + k0 + 8);
      v8h blo = *(const v8h*)(bBase + k0);
      v8h bhi = *(const v8h*)(bBase + k0 + 8);
      acc = __builtin_amdgcn_wmma_f32_16x16x32_f16(
          false, cat8(alo, ahi), false, cat8(blo, bhi),
          (short)0, acc, false, false);
    }

    // LIF1: reset from previous mem, leak+integrate, spike on (mem-1)>0.
    const size_t oidx = ((size_t)t * SNN_B + (size_t)(b0 + hi16 * 8)) * SNN_NH +
                        (size_t)(h0 + wave * 16 + l16);
    float* out = spk_out + oidx;
    _Float16* outh = spkh_out + oidx;
#pragma unroll
    for (int r = 0; r < 8; ++r) {
      float m = mem[r];
      const float reset = (m > 1.0f) ? 1.0f : 0.0f;
      m = beta1 * m + acc[r] - reset;
      mem[r] = m;
      const float s = (m > 1.0f) ? 1.0f : 0.0f;
      out[(size_t)r * SNN_NH] = s;
      if (emit_f16) outh[(size_t)r * SNN_NH] = (_Float16)s;
    }
  }
}

// --------------------------- kernel 2 (fallback) ---------------------------
__global__ void __launch_bounds__(256)
snn_fc2(const float* __restrict__ spk, const float* __restrict__ W2,
        float* __restrict__ cur2) {
  extern __shared__ _Float16 smem2[];
  _Float16* w2s = smem2;                      // [64][NHS_STRIDE] f16
  _Float16* as  = smem2 + 64 * NHS_STRIDE;    // [32][NHS_STRIDE] f16

  const int tid = threadIdx.x;
  const int r0  = blockIdx.x * 32;

  {
    const int row  = tid >> 2;                // 0..63
    const int col0 = (tid & 3) * 256;
    const float* src = W2 + (size_t)row * SNN_NH + col0;
    _Float16* dst = w2s + row * NHS_STRIDE + col0;
#pragma unroll 8
    for (int c = 0; c < 256; c += 4) {
      float4 v = *(const float4*)(src + c);
      v4h h = {(_Float16)v.x, (_Float16)v.y, (_Float16)v.z, (_Float16)v.w};
      *(v4h*)(dst + c) = h;
    }
  }
  {
    const int row  = tid >> 3;                // 0..31
    const int col0 = (tid & 7) * 128;
    const float* src = spk + (size_t)(r0 + row) * SNN_NH + col0;
    _Float16* dst = as + row * NHS_STRIDE + col0;
#pragma unroll 8
    for (int c = 0; c < 128; c += 4) {
      float4 v = *(const float4*)(src + c);
      v4h h = {(_Float16)v.x, (_Float16)v.y, (_Float16)v.z, (_Float16)v.w};
      *(v4h*)(dst + c) = h;
    }
  }
  __syncthreads();

  const int wave = tid >> 5;
  const int lane = tid & 31;
  const int hi16 = lane >> 4;
  const int l16  = lane & 15;
  const int mt   = wave & 1;
  const int nt   = wave >> 1;

  const _Float16* aBase = as  + (mt * 16 + l16) * NHS_STRIDE + hi16 * 16;
  const _Float16* bBase = w2s + (nt * 16 + l16) * NHS_STRIDE + hi16 * 16;

  v8f acc = {};
#pragma unroll 8
  for (int k0 = 0; k0 < SNN_NH; k0 += 32) {
    v8h alo = *(const v8h*)(aBase + k0);
    v8h ahi = *(const v8h*)(aBase + k0 + 8);
    v8h blo = *(const v8h*)(bBase + k0);
    v8h bhi = *(const v8h*)(bBase + k0 + 8);
    acc = __builtin_amdgcn_wmma_f32_16x16x32_f16(
        false, cat8(alo, ahi), false, cat8(blo, bhi),
        (short)0, acc, false, false);
  }

  float* out = cur2 + (size_t)(r0 + mt * 16 + hi16 * 8) * SNN_NO +
               (size_t)(nt * 16 + l16);
#pragma unroll
  for (int r = 0; r < 8; ++r) out[(size_t)r * SNN_NO] = acc[r];
}

// ---------------------------- kernel 2 (async) -----------------------------
// W2 f16 tile staged by one TENSOR_LOAD_TO_LDS (TDM, TENSORcnt) from wave 0;
// spike rows staged per-lane with GLOBAL_LOAD_ASYNC_TO_LDS_B128 (ASYNCcnt).
// Dynamic LDS begins at group-segment offset 0 (no static __shared__ here).
//
// W2 LDS layout is the TDM pad scheme: data_size=2B, tile 1024x64,
// pad_interval=256 dwords (code 7), pad_amount=4 dwords (code 3)
// -> row stride 2080B (1040 halves), +8 halves after half-column 512.
__global__ void __launch_bounds__(256)
snn_fc2_async(const _Float16* __restrict__ spkh,
              const _Float16* __restrict__ w2h, float* __restrict__ cur2) {
  extern __shared__ _Float16 smem3[];
  _Float16* w2s = smem3;                      // [64][W2T_STRIDE] (TDM padded)
  _Float16* as  = smem3 + 64 * W2T_STRIDE;    // [32][NHS_STRIDE]
  const unsigned AS_OFF = 64u * W2T_STRIDE * (unsigned)sizeof(_Float16);

  const int tid = threadIdx.x;
  const int r0  = blockIdx.x * 32;
  const int wave = tid >> 5;

  // --- TDM: one descriptor loads the whole W2 f16 tile into LDS ---------
  if (wave == 0) {
    const unsigned long long ga = (unsigned long long)(__UINTPTR_TYPE__)w2h;
    // D# group0: count=1, lds_addr=0, 57-bit global addr, type=2
    v4u g0 = {1u, 0u, (unsigned)ga,
              (unsigned)((ga >> 32) & 0x1FFFFFFull) | (2u << 30)};
    // D# group1: data_size=1(2B) | pad_enable | pad_interval=7 | pad_amount=3;
    // tensor_dim0=1024, tensor_dim1=64, tile_dim0=1024, tile_dim1=64,
    // tensor_dim0_stride=1024
    v8i g1 = {(int)0x07D10000,
              (int)(1024 << 16),   // barrier_addr=0 | tensor_dim0.lo
              (int)(64 << 16),     // tensor_dim0.hi=0 | tensor_dim1.lo
              (int)(1024 << 16),   // tensor_dim1.hi=0 | tile_dim0
              64,                  // tile_dim1 | tile_dim2=0
              1024,                // tensor_dim0_stride.lo
              0, 0};
    v4i g2 = {0, 0, 0, 0};
    v4i g3 = {0, 0, 0, 0};
    v8i g4 = {0, 0, 0, 0, 0, 0, 0, 0};  // extra group (clang-23 6-arg form)
    __builtin_amdgcn_tensor_load_to_lds(g0, g1, g2, g3, g4, 0);
  }

  // --- async stage 32 spike rows f16: 256B/lane ---------------------------
  {
    const int row  = tid >> 3;
    const int col0 = (tid & 7) * 128;
    const unsigned goff = (unsigned)(
        ((size_t)(r0 + row) * SNN_NH + (size_t)col0) * sizeof(_Float16));
    const unsigned loff =
        AS_OFF + (unsigned)((row * NHS_STRIDE + col0) * sizeof(_Float16));
    AsyncCp<0, 256>::go(loff, goff, (const void*)spkh);
  }
  wait_asynccnt0();
  if (wave == 0) __builtin_amdgcn_s_wait_tensorcnt(0);
  __syncthreads();

  const int lane = tid & 31;
  const int hi16 = lane >> 4;
  const int l16  = lane & 15;
  const int mt   = wave & 1;
  const int nt   = wave >> 1;

  const _Float16* aBase = as + (mt * 16 + l16) * NHS_STRIDE + hi16 * 16;
  const unsigned  bRow  = (unsigned)((nt * 16 + l16) * W2T_STRIDE);

  v8f acc = {};
#pragma unroll 8
  for (int k0 = 0; k0 < SNN_NH; k0 += 32) {
    v8h alo = *(const v8h*)(aBase + k0);
    v8h ahi = *(const v8h*)(aBase + k0 + 8);
    // B fragment: 16 contiguous halves at padded column (bump after col 512)
    const unsigned c = (unsigned)(k0 + hi16 * 16);
    const _Float16* bp = w2s + bRow + c + ((c >> 9) << 3);
    v8h blo = *(const v8h*)bp;
    v8h bhi = *(const v8h*)(bp + 8);
    acc = __builtin_amdgcn_wmma_f32_16x16x32_f16(
        false, cat8(alo, ahi), false, cat8(blo, bhi),
        (short)0, acc, false, false);
  }

  float* out = cur2 + (size_t)(r0 + mt * 16 + hi16 * 8) * SNN_NO +
               (size_t)(nt * 16 + l16);
#pragma unroll
  for (int r = 0; r < 8; ++r) out[(size_t)r * SNN_NO] = acc[r];
}

// -------------------------------- kernel 3 ---------------------------------
__global__ void __launch_bounds__(256)
snn_lif2_scan(float* __restrict__ mem2, const float* __restrict__ beta2p) {
  const int idx = blockIdx.x * blockDim.x + threadIdx.x;  // 0..B*NO-1
  const float beta2 = *beta2p;
  float m = 0.0f;
  for (int t = 0; t < SNN_T; ++t) {
    float* p = mem2 + (size_t)t * (SNN_B * SNN_NO) + idx;
    m = beta2 * m + *p;
    *p = m;
  }
}

// ------------------------------- launcher ----------------------------------
extern "C" void kernel_launch(void* const* d_in, const int* in_sizes, int n_in,
                              void* d_out, int out_size, void* d_ws,
                              size_t ws_size, hipStream_t stream) {
  (void)in_sizes; (void)n_in; (void)out_size;

  const float* x  = (const float*)d_in[0];   // [B,T,NI]
  const float* W1 = (const float*)d_in[1];   // [NH,NI]
  const float* W2 = (const float*)d_in[2];   // [NO,NH]
  const float* b1 = (const float*)d_in[3];   // scalar
  const float* b2 = (const float*)d_in[4];   // scalar

  float* spk  = (float*)d_out;                               // [T,B,NH]
  float* mem2 = spk + (size_t)SNN_T * SNN_B * SNN_NH;        // [T,B,NO]

  const size_t spkh_bytes = (size_t)SNN_T * SNN_B * SNN_NH * sizeof(_Float16);
  const size_t w2h_bytes  = (size_t)SNN_NO * SNN_NH * sizeof(_Float16);
  const bool use_async = (ws_size >= spkh_bytes + w2h_bytes);

  _Float16* spkh = (_Float16*)d_ws;
  _Float16* w2h  = (_Float16*)((char*)d_ws + spkh_bytes);

  if (use_async) {
    snn_w2_to_f16<<<(SNN_NO * SNN_NH) / (256 * 4), 256, 0, stream>>>(W2, w2h);
  }

  const size_t lds1 =
      (size_t)(128 * W1S_STRIDE + 16 * XS_STRIDE) * sizeof(_Float16);
  dim3 g1(SNN_B / 16, SNN_NH / 128);
  snn_fc1_lif1<<<g1, 256, lds1, stream>>>(x, W1, b1, spk, spkh,
                                          use_async ? 1 : 0);

  if (use_async) {
    const size_t lds2a =
        (size_t)(64 * W2T_STRIDE + 32 * NHS_STRIDE) * sizeof(_Float16);
    snn_fc2_async<<<(SNN_T * SNN_B) / 32, 256, lds2a, stream>>>(spkh, w2h,
                                                                mem2);
  } else {
    const size_t lds2 = (size_t)((64 + 32) * NHS_STRIDE) * sizeof(_Float16);
    snn_fc2<<<(SNN_T * SNN_B) / 32, 256, lds2, stream>>>(spk, W2, mem2);
  }

  snn_lif2_scan<<<(SNN_B * SNN_NO) / 256, 256, 0, stream>>>(mem2, b2);
}
